// CaptioningRNN_85521388798686
// MI455X (gfx1250) — matrix operand order repaired and checked by reading, not compiled, not run on hardware
//
#include <hip/hip_runtime.h>
#include <hip/hip_bf16.h>

#define N_BATCH 128
#define T_SEQ   128
#define D_IN    1024
#define H_DIM   1024
#define FOURH   4096

typedef __attribute__((ext_vector_type(16))) __bf16 v16bf;
typedef __attribute__((ext_vector_type(8)))  float  v8f;

union Frag16 { v16bf v; unsigned u[8]; };

__device__ __forceinline__ unsigned bf16_bits(float f) {
  union { float f; unsigned u; } c; c.f = f;
  unsigned u = c.u;
  return (u + 0x7FFFu + ((u >> 16) & 1u)) >> 16;   // round-to-nearest-even
}
__device__ __forceinline__ unsigned pack_bf16x2(float lo, float hi) {
  return bf16_bits(lo) | (bf16_bits(hi) << 16);
}
__device__ __forceinline__ unsigned short bf16_s(float f) {
  return (unsigned short)bf16_bits(f);
}

__device__ __forceinline__ v8f bwmma(const Frag16& a, const Frag16& b, v8f c) {
  // D = A(16x32 bf16) * B(32x16 bf16) + C(16x16 f32)
  return __builtin_amdgcn_wmma_f32_16x16x32_bf16(false, a.v, false, b.v,
                                                 (short)0, c, false, false);
}

// A fragment: 16x32 bf16. lane<16: K 0..7 & 16..23; lane>=16: +8. Two b128 loads.
__device__ __forceinline__ void load_a(Frag16& a, const unsigned short* arow, int k0) {
#pragma unroll
  for (int v = 0; v < 4; ++v) {
    a.u[v]     = *(const unsigned*)(arow + k0 + 2 * v);
    a.u[v + 4] = *(const unsigned*)(arow + k0 + 16 + 2 * v);
  }
}
// B fragment: 32x16 bf16. lane<16: K 0..7 & 16..23; lane>=16: +8. Two b128 loads.
__device__ __forceinline__ void load_b(Frag16& b, const unsigned short* bc, int k0) {
#pragma unroll
  for (int v = 0; v < 4; ++v) {
    b.u[v]     = *(const unsigned*)(bc + k0 + 2 * v);
    b.u[v + 4] = *(const unsigned*)(bc + k0 + 16 + 2 * v);
  }
}

// ---------------------------------------------------------------------------
// Convert f32 weight (K=1024 x N=4096, row-major) -> bf16 transposed [n][k]
// so each B-fragment K-pair is a contiguous dword (b128-vectorizable).
// ---------------------------------------------------------------------------
__global__ __launch_bounds__(256) void wconv_kernel(const float* __restrict__ src,
                                                    unsigned short* __restrict__ dst) {
  int idx = blockIdx.x * 256 + threadIdx.x;     // over 4096*1024
  int col = idx >> 10;
  int k   = idx & 1023;
  dst[col * 1024 + k] = bf16_s(src[k * FOURH + col]);
}

// ---------------------------------------------------------------------------
// Convert x (N*T*D f32, row-major) -> bf16, keeping layout. 4 elems/thread.
// ---------------------------------------------------------------------------
__global__ __launch_bounds__(256) void xconv_kernel(const float* __restrict__ src,
                                                    unsigned* __restrict__ dst2) {
  int idx = blockIdx.x * 256 + threadIdx.x;     // over (N*T*D)/4
  const float4 v = ((const float4*)src)[idx];
  dst2[2 * idx]     = pack_bf16x2(v.x, v.y);
  dst2[2 * idx + 1] = pack_bf16x2(v.z, v.w);
}

// ---------------------------------------------------------------------------
// h0 = c0 = mean over 16 spatial positions of A_flat
// ---------------------------------------------------------------------------
__global__ __launch_bounds__(256) void init_state_kernel(const float* __restrict__ A,
                                                         float* __restrict__ h_cur,
                                                         float* __restrict__ c) {
  int idx = blockIdx.x * 256 + threadIdx.x;     // over N*H
  const float* a = A + idx * 16;
  float s = 0.f;
#pragma unroll
  for (int p = 0; p < 16; ++p) s += a[p];
  s *= (1.f / 16.f);
  h_cur[idx] = s;
  c[idx] = s;
}

// ---------------------------------------------------------------------------
// Precompute xw = x.reshape(N*T, D) @ Wx  (f32 out), pure bf16 WMMA.
// grid (NT/16, 4096/256); 4 waves/block; wave computes 16x64 (4 tiles).
// K-loop fully unrolled; launch_bounds(...,1) lifts the VGPR cap so the
// scheduler can keep more fragment loads in flight (deeper pipeline).
// ---------------------------------------------------------------------------
__global__ __launch_bounds__(128, 1) void xw_gemm_kernel(const unsigned short* __restrict__ xbf,
                                                         const unsigned short* __restrict__ WxT,
                                                         float* __restrict__ xw) {
  const int lane = threadIdx.x & 31;
  const int wave = threadIdx.x >> 5;
  const int m0   = blockIdx.x * 16;
  const int col0 = blockIdx.y * 256 + wave * 64;
  const int mrow = lane & 15;
  const int a_kb = (lane >> 4) * 8;
  const int b_nc = lane & 15;
  const int b_kb = (lane >> 4) * 8;

  v8f acc[4] = {v8f{}, v8f{}, v8f{}, v8f{}};
  const unsigned short* arow = xbf + (m0 + mrow) * D_IN + a_kb;
  const unsigned short* bc[4];
#pragma unroll
  for (int j = 0; j < 4; ++j)
    bc[j] = WxT + (col0 + j * 16 + b_nc) * D_IN + b_kb;

#pragma unroll
  for (int k0 = 0; k0 < D_IN; k0 += 32) {
    Frag16 a;
    load_a(a, arow, k0);
    if ((k0 & 63) == 0 && k0 + 512 < D_IN)
      __builtin_prefetch(arow + k0 + 512, 0, 3);         // 1KB ahead
#pragma unroll
    for (int j = 0; j < 4; ++j) {
      Frag16 b;
      load_b(b, bc[j], k0);
      if ((k0 & 63) == 0 && k0 + 512 < D_IN)
        __builtin_prefetch(bc[j] + k0 + 512, 0, 3);
      acc[j] = bwmma(a, b, acc[j]);
    }
  }

  const int mh = (lane >> 4) * 8;
#pragma unroll
  for (int j = 0; j < 4; ++j)
#pragma unroll
    for (int r = 0; r < 8; ++r)
      xw[(size_t)(m0 + mh + r) * FOURH + col0 + j * 16 + b_nc] = acc[j][r];
}

// ---------------------------------------------------------------------------
// Per-step attention: scores = h . A_flat / 32, softmax(16), attn = A_flat@w.
// Tree-reduced in LDS; converts h_cur and attn to bf16. One block per n.
// ---------------------------------------------------------------------------
__global__ __launch_bounds__(256) void attn_step_kernel(const float* __restrict__ A,
                                                        const float* __restrict__ h_cur,
                                                        unsigned short* __restrict__ hbf,
                                                        unsigned short* __restrict__ attnbf) {
  __shared__ float part[256][16];
  __shared__ float wsm[16];
  const int n = blockIdx.x, tid = threadIdx.x;
  const float* Af = A + (size_t)n * H_DIM * 16;
  const float* h  = h_cur + n * H_DIM;

  float p[16];
#pragma unroll
  for (int q = 0; q < 16; ++q) p[q] = 0.f;
  for (int hh = tid; hh < H_DIM; hh += 256) {
    float hv = h[hh];
    const float* ap = Af + hh * 16;
#pragma unroll
    for (int q = 0; q < 16; ++q) p[q] += hv * ap[q];
  }
#pragma unroll
  for (int q = 0; q < 16; ++q) part[tid][q] = p[q];
  __syncthreads();

  // tree reduction over the 256 partial rows (8 stages)
  for (int s = 128; s >= 1; s >>= 1) {
    if (tid < s * 16) {
      int r = tid >> 4, q = tid & 15;
      part[r][q] += part[r + s][q];
    }
    __syncthreads();
  }
  if (tid == 0) {
    float sc[16];
    float mx = -1e30f;
#pragma unroll
    for (int q = 0; q < 16; ++q) { sc[q] = part[0][q] * 0.03125f; mx = fmaxf(mx, sc[q]); }
    float e[16], den = 0.f;
#pragma unroll
    for (int q = 0; q < 16; ++q) { e[q] = __expf(sc[q] - mx); den += e[q]; }
    float inv = 1.f / den;
#pragma unroll
    for (int q = 0; q < 16; ++q) wsm[q] = e[q] * inv;
  }
  __syncthreads();

  for (int hh = tid; hh < H_DIM; hh += 256) {
    const float* ap = Af + hh * 16;
    float s = 0.f;
#pragma unroll
    for (int q = 0; q < 16; ++q) s += ap[q] * wsm[q];
    attnbf[n * H_DIM + hh] = bf16_s(s);
    hbf[n * H_DIM + hh]    = bf16_s(h[hh]);
  }
}

// ---------------------------------------------------------------------------
// Fused recurrent step: a = h@Wh + attn@Wattn + xw[:,t,:] + b, then LSTM gates.
// grid (128/16, 1024/32); 4 waves/block, wave g owns gate g's 16x32 tile.
// Fully-unrolled K-loops with relaxed VGPR cap for deep load pipelining;
// gate combine through 8KB LDS; writes c, h_cur, out.
// ---------------------------------------------------------------------------
__global__ __launch_bounds__(128, 1) void lstm_step_kernel(
    const unsigned short* __restrict__ hbf, const unsigned short* __restrict__ attnbf,
    const unsigned short* __restrict__ WhT, const unsigned short* __restrict__ WattnT,
    const float* __restrict__ xw, const float* __restrict__ bias,
    float* __restrict__ c, float* __restrict__ h_cur, float* __restrict__ out, int t) {
  __shared__ float gl[4][16][32];
  const int lane = threadIdx.x & 31;
  const int gate = threadIdx.x >> 5;
  const int m0   = blockIdx.x * 16;
  const int colg = gate * H_DIM + blockIdx.y * 32;
  const int mrow = lane & 15;
  const int a_kb = (lane >> 4) * 8;
  const int b_nc = lane & 15;
  const int b_kb = (lane >> 4) * 8;

  v8f acc0 = {}; v8f acc1 = {};
#pragma unroll
  for (int pass = 0; pass < 2; ++pass) {
    const unsigned short* arow =
        (pass == 0 ? hbf : attnbf) + (m0 + mrow) * H_DIM + a_kb;
    const unsigned short* bc0 =
        (pass == 0 ? WhT : WattnT) + (colg + b_nc) * H_DIM + b_kb;
    const unsigned short* bc1 = bc0 + 16 * H_DIM;

#pragma unroll
    for (int k0 = 0; k0 < H_DIM; k0 += 32) {
      Frag16 a, b0, b1;
      load_a(a, arow, k0);
      load_b(b0, bc0, k0);
      load_b(b1, bc1, k0);
      if ((k0 & 63) == 0 && k0 + 512 < H_DIM) {
        __builtin_prefetch(bc0 + k0 + 512, 0, 3);        // hide L2->L0 latency on
        __builtin_prefetch(bc1 + k0 + 512, 0, 3);        // the sequential hot path
      }
      acc0 = bwmma(a, b0, acc0);
      acc1 = bwmma(a, b1, acc1);
    }
  }

  const int mh = (lane >> 4) * 8;
#pragma unroll
  for (int r = 0; r < 8; ++r) {
    int row = m0 + mh + r;
    const float* xwrow = xw + ((size_t)row * T_SEQ + t) * FOURH;
    acc0[r] += xwrow[colg + b_nc]      + bias[colg + b_nc];
    acc1[r] += xwrow[colg + 16 + b_nc] + bias[colg + 16 + b_nc];
    gl[gate][mh + r][b_nc]      = acc0[r];
    gl[gate][mh + r][b_nc + 16] = acc1[r];
  }
  __syncthreads();

  const int hbase = blockIdx.y * 32;
  for (int e = threadIdx.x; e < 16 * 32; e += 128) {
    int m = e >> 5, cc = e & 31;
    int row  = m0 + m;
    int hcol = hbase + cc;
    float ai = gl[0][m][cc], af = gl[1][m][cc], ao = gl[2][m][cc], ag = gl[3][m][cc];
    float ig = 1.f / (1.f + __expf(-ai));
    float fg = 1.f / (1.f + __expf(-af));
    float og = 1.f / (1.f + __expf(-ao));
    float gg = tanhf(ag);
    float cv  = c[row * H_DIM + hcol];
    float ncv = fg * cv + ig * gg;
    float nh  = og * tanhf(ncv);
    c[row * H_DIM + hcol]     = ncv;
    h_cur[row * H_DIM + hcol] = nh;
    out[((size_t)row * T_SEQ + t) * H_DIM + hcol] = nh;
  }
}

// ---------------------------------------------------------------------------
extern "C" void kernel_launch(void* const* d_in, const int* in_sizes, int n_in,
                              void* d_out, int out_size, void* d_ws, size_t ws_size,
                              hipStream_t stream) {
  (void)in_sizes; (void)n_in; (void)out_size; (void)ws_size;
  const float* x     = (const float*)d_in[0];   // (N, T, D)
  const float* A     = (const float*)d_in[1];   // (N, H, 4, 4)
  const float* Wx    = (const float*)d_in[2];   // (D, 4H)
  const float* Wh    = (const float*)d_in[3];   // (H, 4H)
  const float* Wattn = (const float*)d_in[4];   // (H, 4H)
  const float* bias  = (const float*)d_in[5];   // (4H,)
  float* out = (float*)d_out;                   // (N, T, H)

  char* ws = (char*)d_ws;
  size_t off = 0;
  auto carve = [&](size_t bytes) -> void* {
    void* p = ws + off;
    off = (off + bytes + 255) & ~(size_t)255;
    return p;
  };
  unsigned short* WxT    = (unsigned short*)carve((size_t)FOURH * H_DIM * 2);
  unsigned short* WhT    = (unsigned short*)carve((size_t)FOURH * H_DIM * 2);
  unsigned short* WattnT = (unsigned short*)carve((size_t)FOURH * H_DIM * 2);
  unsigned short* xbf    = (unsigned short*)carve((size_t)N_BATCH * T_SEQ * D_IN * 2);
  float*          xw     = (float*)carve((size_t)N_BATCH * T_SEQ * FOURH * 4);
  float*          h_cur  = (float*)carve((size_t)N_BATCH * H_DIM * 4);
  float*          cstate = (float*)carve((size_t)N_BATCH * H_DIM * 4);
  unsigned short* hbf    = (unsigned short*)carve((size_t)N_BATCH * H_DIM * 2);
  unsigned short* attnbf = (unsigned short*)carve((size_t)N_BATCH * H_DIM * 2);

  // One-time preprocessing (bf16 weights total 24MB -> resident in 192MB L2).
  wconv_kernel<<<(FOURH * H_DIM) / 256, 256, 0, stream>>>(Wx, WxT);
  wconv_kernel<<<(FOURH * H_DIM) / 256, 256, 0, stream>>>(Wh, WhT);
  wconv_kernel<<<(FOURH * H_DIM) / 256, 256, 0, stream>>>(Wattn, WattnT);
  xconv_kernel<<<(N_BATCH * T_SEQ * D_IN) / (256 * 4), 256, 0, stream>>>(x, (unsigned*)xbf);
  init_state_kernel<<<(N_BATCH * H_DIM) / 256, 256, 0, stream>>>(A, h_cur, cstate);

  // Hoisted input projection: (N*T, D) @ (D, 4H), fully parallel.
  xw_gemm_kernel<<<dim3((N_BATCH * T_SEQ) / 16, FOURH / 256), 128, 0, stream>>>(xbf, WxT, xw);

  // Sequential recurrence: attention + fused WMMA gate GEMM per step.
  for (int t = 0; t < T_SEQ; ++t) {
    attn_step_kernel<<<N_BATCH, 256, 0, stream>>>(A, h_cur, hbf, attnbf);
    lstm_step_kernel<<<dim3(N_BATCH / 16, H_DIM / 32), 128, 0, stream>>>(
        hbf, attnbf, WhT, WattnT, xw, bias, cstate, h_cur, out, t);
  }
}
